// Deformable_spatemp_atten_network_13477607375420
// MI455X (gfx1250) — matrix operand order repaired and checked by reading, not compiled, not run on hardware
//
#include <hip/hip_runtime.h>

typedef __attribute__((ext_vector_type(16))) _Float16 v16h;
typedef __attribute__((ext_vector_type(8)))  float    v8f;

// ---------------------------------------------------------------------------
// WMMA fragment index helpers (CDNA5 ISA 7.12.2 layouts, wave32)
//   A (16x32 f16): lane -> M = lane&15 ; K subgroup by lane half
//   B (32x16 f16): lane -> N = lane&15 ; K by lane half, same order as A
//   C/D (16x16 f32, v8f): lane -> N = lane&15 ; reg r -> M = r + 8*(lane>>4)
// ---------------------------------------------------------------------------
__device__ __forceinline__ int a_k_of(int lane, int i) {
  int base = (lane >> 4) << 3;          // 0 or 8
  int vg = i >> 1, pos = i & 1;
  return (vg < 4) ? (base + (vg << 1) + pos)
                  : (16 + base + ((vg - 4) << 1) + pos);
}
__device__ __forceinline__ int reflect1(int i, int n) {
  return i < 0 ? -i : (i >= n ? 2 * n - 2 - i : i);
}

// ---------------------------------------------------------------------------
// Implicit-GEMM convolution. One wave computes a 16(Cout) x 64(spatial) tile:
// 4 WMMA accumulators reuse a single A (weight) fragment per K-step.
// mode 0: 3x3 s1 reflect-pad1      mode 1: 4x4 s2 reflect-pad1
// mode 2: 1x1                      mode 3: transposed 4x4 s2 p1 (zero-stuffed)
// mode 4: 8x8 s8 p0 (offset conv)
// ---------------------------------------------------------------------------
__global__ void conv_wmma_k(const float* __restrict__ X, const float* __restrict__ Wt,
                            const float* __restrict__ bias, float* __restrict__ Y,
                            int Cin, int Hin, int Win, int Cout, int Ho, int Wo,
                            int K, int mode) {
  const int lane = threadIdx.x & 31;
  const int wave = threadIdx.x >> 5;
  const int frame = blockIdx.z;
  const int m0 = blockIdx.y * 16;
  const int n0 = (blockIdx.x * 4 + wave) * 64;     // 64 spatial cols per wave
  const int HW = Ho * Wo;
  if (n0 >= HW) return;                            // wave-uniform
  const float* Xf = X + (long long)frame * Cin * Hin * Win;
  v8f acc0 = {}, acc1 = {}, acc2 = {}, acc3 = {};
  const int mA = m0 + (lane & 15);
  const int kbl = (lane & 15) + ((lane >> 4) << 4);
  for (int k0 = 0; k0 < K; k0 += 32) {
    // ---- A fragment (weights), loaded once, reused by 4 WMMAs ----
    v16h a;
#pragma unroll
    for (int i = 0; i < 16; i++) {
      int k = k0 + a_k_of(lane, i);
      float v = (mA < Cout && k < K) ? Wt[(long long)mA * K + k] : 0.f;
      a[i] = (_Float16)v;
    }
    if (mA < Cout && k0 + 32 < K)                  // next K-step weight row
      __builtin_prefetch(Wt + (long long)mA * K + k0 + 32, 0, 1);

#pragma unroll
    for (int j = 0; j < 4; j++) {
      int nb = n0 + j * 16;
      if (nb >= HW) continue;                      // wave-uniform
      v16h b;
#pragma unroll
      for (int i = 0; i < 16; i++) {
        int n = nb + (lane & 15);
        int kb = k0 + a_k_of(lane, i);
        int ci = 0, kh = 0, kw = 0;
        bool kok = (kb < K);
        if (kok) {
          if (mode == 0)      { ci = kb / 9;  int r = kb % 9;  kh = r / 3; kw = r % 3; }
          else if (mode == 1 || mode == 3) { ci = kb >> 4; int r = kb & 15; kh = r >> 2; kw = r & 3; }
          else if (mode == 2) { ci = kb; }
          else                { ci = kb >> 6; int r = kb & 63; kh = r >> 3; kw = r & 7; }
        }
        const float* Xc = Xf + (long long)ci * Hin * Win;
        float v = 0.f;
        if (kok && n < HW) {
          int h = n / Wo, w = n % Wo;
          if (mode == 0) {
            v = Xc[reflect1(h - 1 + kh, Hin) * Win + reflect1(w - 1 + kw, Win)];
          } else if (mode == 1) {
            v = Xc[reflect1(2 * h - 1 + kh, Hin) * Win + reflect1(2 * w - 1 + kw, Win)];
          } else if (mode == 2) {
            v = Xc[n];
          } else if (mode == 3) {
            int jh = h + kh - 2, jw = w + kw - 2;
            if (jh >= 0 && jw >= 0 && !(jh & 1) && !(jw & 1)) {
              int sh = jh >> 1, sw = jw >> 1;
              if (sh < Hin && sw < Win) v = Xc[sh * Win + sw];
            }
          } else {
            v = Xc[(8 * h + kh) * Win + (8 * w + kw)];
          }
        }
        b[i] = (_Float16)v;
      }
      if (j == 0) acc0 = __builtin_amdgcn_wmma_f32_16x16x32_f16(false, a, false, b, (short)0, acc0, false, false);
      else if (j == 1) acc1 = __builtin_amdgcn_wmma_f32_16x16x32_f16(false, a, false, b, (short)0, acc1, false, false);
      else if (j == 2) acc2 = __builtin_amdgcn_wmma_f32_16x16x32_f16(false, a, false, b, (short)0, acc2, false, false);
      else acc3 = __builtin_amdgcn_wmma_f32_16x16x32_f16(false, a, false, b, (short)0, acc3, false, false);
    }
  }
  float* Yf = Y + (long long)frame * Cout * HW;
#pragma unroll
  for (int j = 0; j < 4; j++) {
    int n = n0 + j * 16 + (lane & 15);
    if (n >= HW) continue;
    v8f acc = (j == 0) ? acc0 : (j == 1) ? acc1 : (j == 2) ? acc2 : acc3;
#pragma unroll
    for (int r = 0; r < 8; r++) {
      int m = m0 + r + 8 * (lane >> 4);
      if (m < Cout) Yf[(long long)m * HW + n] = acc[r] + bias[m];
    }
  }
}

// ---------------------------------------------------------------------------
// Batched-strided generic GEMM:  C = act(alpha * A@B + bias)
// One wave computes a 16(M) x 64(N) tile (4 accumulators, shared A fragment).
// tA/tB/tC: 0 -> [row, col] row-major ; 1 -> transposed indexing
// biasMode: 0 none, 1 per-row(m), 2 per-col(n)
// batch z: b0 = z/nb1, b1 = z%nb1; operand offset = b0*s?0 + b1*s?1
// ---------------------------------------------------------------------------
__global__ void gemm_wmma_k(const float* __restrict__ A, const float* __restrict__ Bm,
                            const float* __restrict__ bias, float* __restrict__ Cm,
                            int M, int N, int K, int lda, int ldb, int ldc,
                            int tA, int tB, int tC, int biasMode, int act, float alpha,
                            int nb1, long long sA0, long long sA1, long long sB0,
                            long long sB1, long long sC0, long long sC1) {
  int b0 = blockIdx.z / nb1, b1 = blockIdx.z % nb1;
  A  += b0 * sA0 + b1 * sA1;
  Bm += b0 * sB0 + b1 * sB1;
  Cm += b0 * sC0 + b1 * sC1;
  int lane = threadIdx.x & 31, wave = threadIdx.x >> 5;
  int m0 = blockIdx.y * 16;
  int n0 = (blockIdx.x * 4 + wave) * 64;
  if (n0 >= N || m0 >= M) return;                  // wave-uniform
  v8f acc0 = {}, acc1 = {}, acc2 = {}, acc3 = {};
  int ma = m0 + (lane & 15);
  int kbl = (lane & 15) + ((lane >> 4) << 4);
  for (int k0 = 0; k0 < K; k0 += 32) {
    v16h a;
#pragma unroll
    for (int i = 0; i < 16; i++) {
      int k = k0 + a_k_of(lane, i);
      float v = 0.f;
      if (ma < M && k < K) v = tA ? A[(long long)k * lda + ma] : A[(long long)ma * lda + k];
      a[i] = (_Float16)v;
    }

#pragma unroll
    for (int j = 0; j < 4; j++) {
      int nb = n0 + j * 16;
      if (nb >= N) continue;                       // wave-uniform
      v16h b;
#pragma unroll
      for (int i = 0; i < 16; i++) {
        int n = nb + (lane & 15);
        int kb = k0 + a_k_of(lane, i);
        float v = 0.f;
        if (kb < K && n < N) v = tB ? Bm[(long long)n * ldb + kb] : Bm[(long long)kb * ldb + n];
        b[i] = (_Float16)v;
      }
      if (j == 0) acc0 = __builtin_amdgcn_wmma_f32_16x16x32_f16(false, a, false, b, (short)0, acc0, false, false);
      else if (j == 1) acc1 = __builtin_amdgcn_wmma_f32_16x16x32_f16(false, a, false, b, (short)0, acc1, false, false);
      else if (j == 2) acc2 = __builtin_amdgcn_wmma_f32_16x16x32_f16(false, a, false, b, (short)0, acc2, false, false);
      else acc3 = __builtin_amdgcn_wmma_f32_16x16x32_f16(false, a, false, b, (short)0, acc3, false, false);
    }
  }
#pragma unroll
  for (int j = 0; j < 4; j++) {
    int n = n0 + j * 16 + (lane & 15);
    if (n >= N) continue;
    v8f acc = (j == 0) ? acc0 : (j == 1) ? acc1 : (j == 2) ? acc2 : acc3;
#pragma unroll
    for (int r = 0; r < 8; r++) {
      int m = m0 + r + 8 * (lane >> 4);
      if (m < M) {
        float v = acc[r] * alpha;
        if (biasMode == 1) v += bias[m];
        else if (biasMode == 2) v += bias[n];
        if (act) v = fmaxf(v, 0.f);
        if (tC) Cm[(long long)n * ldc + m] = v;
        else    Cm[(long long)m * ldc + n] = v;
      }
    }
  }
}

// --------------------------- normalization ---------------------------------
__global__ void group_norm_k(const float* __restrict__ X, float* __restrict__ Y,
                             int C, int HW, int groups) {
  int g = blockIdx.x % groups, frame = blockIdx.x / groups;
  int Cg = C / groups;
  long long base = (long long)frame * C * HW + (long long)g * Cg * HW;
  int cnt = Cg * HW;
  __shared__ float rs[256], rq[256];
  float s = 0.f, q = 0.f;
  for (int i = threadIdx.x; i < cnt; i += blockDim.x) {
    float v = X[base + i]; s += v; q += v * v;
  }
  rs[threadIdx.x] = s; rq[threadIdx.x] = q; __syncthreads();
  for (int st = 128; st > 0; st >>= 1) {
    if (threadIdx.x < st) { rs[threadIdx.x] += rs[threadIdx.x + st]; rq[threadIdx.x] += rq[threadIdx.x + st]; }
    __syncthreads();
  }
  float mean = rs[0] / cnt;
  float inv = rsqrtf(rq[0] / cnt - mean * mean + 1e-5f);
  for (int i = threadIdx.x; i < cnt; i += blockDim.x)
    Y[base + i] = fmaxf((X[base + i] - mean) * inv, 0.f);
}

__global__ void batch_norm_k(const float* __restrict__ X, float* __restrict__ Y,
                             int Nb, int C, int HW, int cOff, int Ctot) {
  int c = blockIdx.x;
  __shared__ float rs[256], rq[256];
  long long cnt = (long long)Nb * HW;
  float s = 0.f, q = 0.f;
  for (long long i = threadIdx.x; i < cnt; i += blockDim.x) {
    int n = (int)(i / HW), j = (int)(i % HW);
    float v = X[((long long)n * C + c) * HW + j]; s += v; q += v * v;
  }
  rs[threadIdx.x] = s; rq[threadIdx.x] = q; __syncthreads();
  for (int st = 128; st > 0; st >>= 1) {
    if (threadIdx.x < st) { rs[threadIdx.x] += rs[threadIdx.x + st]; rq[threadIdx.x] += rq[threadIdx.x + st]; }
    __syncthreads();
  }
  float mean = rs[0] / cnt;
  float inv = rsqrtf(rq[0] / cnt - mean * mean + 1e-5f);
  for (long long i = threadIdx.x; i < cnt; i += blockDim.x) {
    int n = (int)(i / HW), j = (int)(i % HW);
    float v = (X[((long long)n * C + c) * HW + j] - mean) * inv;
    Y[((long long)n * Ctot + cOff + c) * HW + j] = fmaxf(v, 0.f);
  }
}

// --------------------------- elementwise -----------------------------------
__global__ void add_k(float* __restrict__ O, const float* __restrict__ A,
                      const float* __restrict__ B, long long n) {
  long long i = (long long)blockIdx.x * blockDim.x + threadIdx.x;
  if (i < n) O[i] = A[i] + B[i];
}

__global__ void pos_encode_k(const int* __restrict__ bp, float* __restrict__ pe, int T) {
  int t = blockIdx.x, j = threadIdx.x;          // 64 threads
  if (t >= T) return;
  int i = j & 15;
  float denom = powf(10000.f, 2.f * (float)(i / 2) / 16.f);
  float ang = (float)bp[t] / denom;             // batch_positions[0] row
  pe[t * 64 + j] = (i & 1) ? cosf(ang) : sinf(ang);
}

__global__ void add_pos_k(float* __restrict__ X, const float* __restrict__ pe,
                          int T, int C, int HW, long long total) {
  long long i = (long long)blockIdx.x * blockDim.x + threadIdx.x;
  if (i >= total) return;
  long long chw = (long long)C * HW;
  int f = (int)(i / chw);
  int c = (int)((i % chw) / HW);
  X[i] += pe[(f % T) * C + c];
}

// --------------------------- DTA pieces ------------------------------------
__global__ void dta_sample_k(const float* __restrict__ X, const float* __restrict__ OFF,
                             float* __restrict__ S, int C, int H, int W, int Hg, int Wg) {
  int f = blockIdx.y, g = blockIdx.x;
  int Ng = Hg * Wg;
  int hg = g / Wg, wg = g % Wg;
  const float* of = OFF + (long long)f * 2 * Ng;
  float ys = (hg + 0.5f) * 8.f - 0.5f + tanhf(of[g]) * 8.f;
  float xs = (wg + 0.5f) * 8.f - 0.5f + tanhf(of[Ng + g]) * 8.f;
  ys = fminf(fmaxf(ys, 0.f), (float)(H - 1));
  xs = fminf(fmaxf(xs, 0.f), (float)(W - 1));
  int y0 = (int)floorf(ys), x0 = (int)floorf(xs);
  float wy = ys - y0, wx = xs - x0;
  y0 = min(max(y0, 0), H - 1); x0 = min(max(x0, 0), W - 1);
  int y1 = min(y0 + 1, H - 1), x1 = min(x0 + 1, W - 1);
  const float* Xf = X + (long long)f * C * H * W;
  for (int c = threadIdx.x; c < C; c += blockDim.x) {
    const float* Xc = Xf + (long long)c * H * W;
    float v = Xc[y0 * W + x0] * (1 - wy) * (1 - wx) + Xc[y0 * W + x1] * (1 - wy) * wx +
              Xc[y1 * W + x0] * wy * (1 - wx) + Xc[y1 * W + x1] * wy * wx;
    S[((long long)f * Ng + g) * C + c] = v;
  }
}

__global__ void softmax_rows_k(float* __restrict__ X, long long rows, int n) {
  long long r = (long long)blockIdx.x * blockDim.x + threadIdx.x;
  if (r >= rows) return;
  float* p = X + r * n;
  float mx = -1e30f;
  for (int i = 0; i < n; i++) mx = fmaxf(mx, p[i]);
  float s = 0.f;
  for (int i = 0; i < n; i++) { float e = __expf(p[i] - mx); p[i] = e; s += e; }
  float inv = 1.f / s;
  for (int i = 0; i < n; i++) p[i] *= inv;
}

// --------------------------- time aggregation -------------------------------
__global__ void spatial_avg_k(const float* __restrict__ X, float* __restrict__ A, int HW) {
  __shared__ float red[256];
  long long base = (long long)blockIdx.x * HW;
  float s = 0.f;
  for (int i = threadIdx.x; i < HW; i += blockDim.x) s += X[base + i];
  red[threadIdx.x] = s; __syncthreads();
  for (int st = 128; st > 0; st >>= 1) {
    if (threadIdx.x < st) red[threadIdx.x] += red[threadIdx.x + st];
    __syncthreads();
  }
  if (threadIdx.x == 0) A[blockIdx.x] = red[0] / HW;
}

__global__ void agg_mlp_k(const float* __restrict__ A, const float* __restrict__ w1,
                          const float* __restrict__ b1, const float* __restrict__ w2,
                          const float* __restrict__ b2, float* __restrict__ wgt, int C) {
  int f = blockIdx.x, j = threadIdx.x;          // blockDim = 64
  int Ch = C / 2;
  __shared__ float red[64];
  float part = 0.f;
  if (j < Ch) {
    float h = b1[j];
    for (int c = 0; c < C; c++) h += A[(long long)f * C + c] * w1[c * Ch + j];
    part = fmaxf(h, 0.f) * w2[j];
  }
  red[j] = part; __syncthreads();
  for (int st = 32; st > 0; st >>= 1) {
    if (j < st) red[j] += red[j + st];
    __syncthreads();
  }
  if (j == 0) wgt[f] = 1.f / (1.f + __expf(-(red[0] + b2[0])));
}

__global__ void time_wsum_k(const float* __restrict__ X, const float* __restrict__ w,
                            float* __restrict__ Y, int T, int C, int HW, long long total) {
  long long i = (long long)blockIdx.x * blockDim.x + threadIdx.x;
  if (i >= total) return;
  long long chw = (long long)C * HW;
  int b = (int)(i / chw);
  long long j = i % chw;
  float s = 0.f;
  for (int t = 0; t < T; t++) s += X[(long long)(b * T + t) * chw + j] * w[b * T + t];
  Y[i] = s;
}

// ===========================================================================
// Host orchestration
// ===========================================================================
static inline int cdiv(int a, int b) { return (a + b - 1) / b; }

extern "C" void kernel_launch(void* const* d_in, const int* in_sizes, int n_in,
                              void* d_out, int out_size, void* d_ws, size_t ws_size,
                              hipStream_t stream) {
  (void)in_sizes; (void)n_in; (void)out_size; (void)ws_size;
  const float* x  = (const float*)d_in[0];      // (2,12,10,64,64)
  const int*   bp = (const int*)d_in[1];        // (2,12) -> row 0 used
  // params flattened: sorted dict keys / list-order leaves
  const float* P[114];
  for (int i = 0; i < 114; i++) P[i] = (const float*)d_in[2 + i];
  // agg{0..3}: +0 b1 +1 b2 +2 w1 +3 w2          (base 0,4,8,12)
  // down{0..2}: +0 c1w +1 c1b +2 c2w +3 c2b +4 dw +5 db  (base 16,22,28)
  // dta{0..3}: +0 bk +1 bo +2 boff +3 bq +4 bv +5 wk +6 wo +7 woff +8 wq +9 wv (34,44,54,64)
  // head{0..2}: +0 w0 +1 b0 +2 w1 +3 b1        (74,78,82)
  // in_conv: 86 w0, 87 b0, 88 w1, 89 b1
  // up{0..2}: +0 bskip +1 bup +2 c1w +3 c1b +4 c2w +5 c2b +6 wskip +7 wup (90,98,106)

  float* ws = (float*)d_ws;
  long long off = 0;
  auto alloc = [&](long long n) { float* p = ws + off; off += n; return p; };
  float* CUR0 = alloc(6300000);
  float* CUR1 = alloc(1600000);
  float* CUR2 = alloc(400000);
  float* CUR3 = alloc(200000);
  float* F0 = alloc(524288), *F1 = alloc(131072), *F2 = alloc(32768), *F3 = alloc(16384);
  float* TA = alloc(6300000), *TB = alloc(6300000);
  float* Q  = alloc(6300000), *OB = alloc(6300000), *DTO = alloc(6300000);
  float* SCb = alloc(25200000);
  float* Sb = alloc(200000), *KKb = alloc(200000), *VVb = alloc(200000);
  float* OFFb = alloc(8192), *AVb = alloc(8192), *WGb = alloc(64), *PEb = alloc(1024);
  float* CATb = alloc(1100000);
  float* UA = alloc(600000), *UB = alloc(600000);
  float* UPo1 = alloc(32768), *UPo2 = alloc(131072), *UPo3 = alloc(524288);

  auto conv = [&](const float* X, const float* Wt, const float* b, float* Y, int Nf,
                  int Cin, int Hin, int Win, int Cout, int Ho, int Wo, int mode) {
    int K = (mode == 0) ? Cin * 9 : (mode == 2) ? Cin : (mode == 4) ? Cin * 64 : Cin * 16;
    dim3 g(cdiv(Ho * Wo, 256), cdiv(Cout, 16), Nf);   // 4 waves x 64 cols
    conv_wmma_k<<<g, 128, 0, stream>>>(X, Wt, b, Y, Cin, Hin, Win, Cout, Ho, Wo, K, mode);
  };
  auto gnorm = [&](float* X, int Nf, int C, int HW) {
    group_norm_k<<<Nf * 4, 256, 0, stream>>>(X, X, C, HW, 4);
  };
  auto bnorm = [&](const float* X, float* Y, int Nb, int C, int HW, int cOff, int Ctot) {
    batch_norm_k<<<C, 256, 0, stream>>>(X, Y, Nb, C, HW, cOff, Ctot);
  };
  auto vadd = [&](float* O, const float* A, const float* B, long long n) {
    add_k<<<(int)((n + 255) / 256), 256, 0, stream>>>(O, A, B, n);
  };
  auto gemm = [&](const float* A, int lda, int tA, const float* B, int ldb, int tB,
                  const float* bias, int biasMode, float* C, int ldc, int tC,
                  int M, int N, int K, float alpha, int Z, int nb1, long long sA0,
                  long long sA1, long long sB0, long long sB1, long long sC0,
                  long long sC1) {
    dim3 g(cdiv(N, 256), cdiv(M, 16), Z);
    gemm_wmma_k<<<g, 128, 0, stream>>>(A, B, bias, C, M, N, K, lda, ldb, ldc, tA, tB, tC,
                                       biasMode, 0, alpha, nb1, sA0, sA1, sB0, sB1, sC0, sC1);
  };

  // -------------------- in_conv + positional encoding ----------------------
  conv(x, P[86], P[87], TA, 24, 10, 64, 64, 64, 64, 64, 0);
  gnorm(TA, 24, 64, 4096);
  conv(TA, P[88], P[89], CUR0, 24, 64, 64, 64, 64, 64, 64, 0);
  gnorm(CUR0, 24, 64, 4096);
  pos_encode_k<<<12, 64, 0, stream>>>(bp, PEb, 12);
  add_pos_k<<<(int)((24LL * 64 * 4096 + 255) / 256), 256, 0, stream>>>(
      CUR0, PEb, 12, 64, 4096, 24LL * 64 * 4096);

  // -------------------- DTA + time aggregation per level -------------------
  auto run_dta_agg = [&](const float* XB, int dB, int aB, int C, int H, int W, float* FOUT) {
    int Nf = 24, HW = H * W, Hg = H / 8, Wg = W / 8, Ng = Hg * Wg, NH = 4, dh = C / 4;
    conv(XB, P[dB + 8], P[dB + 3], Q, Nf, C, H, W, C, H, W, 2);            // q = wq*x
    conv(Q, P[dB + 7], P[dB + 2], OFFb, Nf, C, H, W, 2, Hg, Wg, 4);        // offsets
    dta_sample_k<<<dim3(Ng, Nf), (C < 256 ? C : 256), 0, stream>>>(XB, OFFb, Sb, C, H, W, Hg, Wg);
    // kk = s@wk + bk ; vv = s@wv + bv   (batched over frames)
    gemm(Sb, C, 0, P[dB + 5], C, 0, P[dB + 0], 2, KKb, C, 0, Ng, C, C, 1.f,
         Nf, 1, (long long)Ng * C, 0, 0, 0, (long long)Ng * C, 0);
    gemm(Sb, C, 0, P[dB + 9], C, 0, P[dB + 4], 2, VVb, C, 0, Ng, C, C, 1.f,
         Nf, 1, (long long)Ng * C, 0, 0, 0, (long long)Ng * C, 0);
    // scores[frame,h][HW,Ng] = (q_h^T @ k_h^T) / sqrt(dh)
    gemm(Q, HW, 1, KKb, C, 1, nullptr, 0, SCb, Ng, 0, HW, Ng, dh,
         rsqrtf((float)dh), Nf * NH, NH,
         (long long)C * HW, (long long)dh * HW, (long long)Ng * C, dh,
         (long long)NH * HW * Ng, (long long)HW * Ng);
    long long rows = (long long)Nf * NH * HW;
    softmax_rows_k<<<(int)((rows + 255) / 256), 256, 0, stream>>>(SCb, rows, Ng);
    // o_h = attn @ v_h, stored channel-major into OB (Nf,C,HW)
    gemm(SCb, Ng, 0, VVb, C, 0, nullptr, 0, OB, HW, 1, HW, dh, Ng, 1.f,
         Nf * NH, NH, (long long)NH * HW * Ng, (long long)HW * Ng,
         (long long)Ng * C, dh, (long long)C * HW, (long long)dh * HW);
    conv(OB, P[dB + 6], P[dB + 1], TA, Nf, C, H, W, C, H, W, 2);           // wo proj
    vadd(DTO, XB, TA, (long long)Nf * C * HW);                             // residual
    // time aggregation
    spatial_avg_k<<<Nf * C, 256, 0, stream>>>(DTO, AVb, HW);
    agg_mlp_k<<<Nf, 64, 0, stream>>>(AVb, P[aB + 2], P[aB + 0], P[aB + 3], P[aB + 1], WGb, C);
    long long tot = 2LL * C * HW;
    time_wsum_k<<<(int)((tot + 255) / 256), 256, 0, stream>>>(DTO, WGb, FOUT, 12, C, HW, tot);
  };

  auto run_down = [&](const float* XB, int pB, int Cin, int Cout, int H, int W, float* OUT) {
    int Nf = 24, Ho = H / 2, Wo = W / 2, HW = Ho * Wo;
    conv(XB, P[pB + 4], P[pB + 5], TA, Nf, Cin, H, W, Cin, Ho, Wo, 1);
    gnorm(TA, Nf, Cin, HW);
    conv(TA, P[pB + 0], P[pB + 1], TB, Nf, Cin, Ho, Wo, Cout, Ho, Wo, 0);
    gnorm(TB, Nf, Cout, HW);
    conv(TB, P[pB + 2], P[pB + 3], TA, Nf, Cout, Ho, Wo, Cout, Ho, Wo, 0);
    gnorm(TA, Nf, Cout, HW);
    vadd(OUT, TB, TA, (long long)Nf * Cout * HW);
  };

  run_dta_agg(CUR0, 34, 0, 64, 64, 64, F0);
  run_down(CUR0, 16, 64, 64, 64, 64, CUR1);
  run_dta_agg(CUR1, 44, 4, 64, 32, 32, F1);
  run_down(CUR1, 22, 64, 64, 32, 32, CUR2);
  run_dta_agg(CUR2, 54, 8, 64, 16, 16, F2);
  run_down(CUR2, 28, 64, 128, 16, 16, CUR3);
  run_dta_agg(CUR3, 64, 12, 128, 8, 8, F3);

  // -------------------- decoder ---------------------------------------------
  auto run_up = [&](const float* UPin, const float* SKIP, int pB, int din, int dout,
                    int dskip, int h, int w, float* OUT) {
    int Ho = 2 * h, Wo = 2 * w, HW = Ho * Wo, Ctot = dout + dskip;
    conv(UPin, P[pB + 7], P[pB + 1], UA, 2, din, h, w, dout, Ho, Wo, 3);   // tconv
    bnorm(UA, CATb, 2, dout, HW, 0, Ctot);
    conv(SKIP, P[pB + 6], P[pB + 0], UA, 2, dskip, Ho, Wo, dskip, Ho, Wo, 2);
    bnorm(UA, CATb, 2, dskip, HW, dout, Ctot);
    conv(CATb, P[pB + 2], P[pB + 3], UA, 2, Ctot, Ho, Wo, dout, Ho, Wo, 0);
    bnorm(UA, UA, 2, dout, HW, 0, dout);
    conv(UA, P[pB + 4], P[pB + 5], UB, 2, dout, Ho, Wo, dout, Ho, Wo, 0);
    bnorm(UB, UB, 2, dout, HW, 0, dout);
    vadd(OUT, UA, UB, 2LL * dout * HW);
  };
  run_up(F3, F2, 90, 128, 64, 64, 8, 8, UPo1);
  run_up(UPo1, F1, 98, 64, 64, 64, 16, 16, UPo2);
  run_up(UPo2, F0, 106, 64, 64, 64, 32, 32, UPo3);

  // -------------------- heads (out_g=head2, out_1=head0, out_2=head1) -------
  float* outG = (float*)d_out;
  float* out1 = outG + 2LL * 52 * 4096;
  float* out2 = out1 + 2LL * 6 * 4096;
  auto run_head = [&](int pB, int ncls, float* OUTp) {
    conv(UPo3, P[pB + 0], P[pB + 1], UA, 2, 64, 64, 64, 64, 64, 64, 0);
    bnorm(UA, UA, 2, 64, 4096, 0, 64);
    conv(UA, P[pB + 2], P[pB + 3], UB, 2, 64, 64, 64, ncls, 64, 64, 0);
    bnorm(UB, OUTp, 2, ncls, 4096, 0, ncls);
  };
  run_head(82, 52, outG);
  run_head(74, 6, out1);
  run_head(78, 20, out2);
}
